// DeformConv_67491116089498
// MI455X (gfx1250) — matrix operand order repaired and checked by reading, not compiled, not run on hardware
//
#include <hip/hip_runtime.h>
#include <hip/hip_bf16.h>

typedef __attribute__((ext_vector_type(16))) _Float16 v16h;
typedef __attribute__((ext_vector_type(8)))  _Float16 v8h;
typedef __attribute__((ext_vector_type(2)))  _Float16 v2h;
typedef __attribute__((ext_vector_type(8)))  float    v8f;

#define CIN   256
#define COUT  256
#define Hdim  96
#define Wdim  96
#define HW    (Hdim*Wdim)          // 9216
#define KDIM  (CIN*9)              // 2304
#define NIMG  4

// ---------------------------------------------------------------------------
// Async global->LDS (CDNA5): data goes straight to LDS, tracked by ASYNCcnt.
// GV form: vdst = LDS byte address (wave-relative), vaddr = 64-bit global addr.
// ---------------------------------------------------------------------------
__device__ __forceinline__ void async_ld128(unsigned lds_off, const void* gptr) {
    asm volatile("global_load_async_to_lds_b128 %0, %1, off"
                 :: "v"(lds_off), "v"((unsigned long long)(size_t)gptr)
                 : "memory");
}
__device__ __forceinline__ void wait_async0() {
    asm volatile("s_wait_asynccnt 0x0" ::: "memory");
}

// ---------------------------------------------------------------------------
// Pack main-conv weights: wpk[o][k*256+c] = (f16) w[o][c][k]   (w is [256][256][9])
// ---------------------------------------------------------------------------
__global__ __launch_bounds__(256) void pack_w_kernel(const float* __restrict__ w,
                                                     _Float16* __restrict__ wpk) {
    int idx = blockIdx.x * 256 + threadIdx.x;          // 0 .. 256*2304-1
    if (idx >= COUT * KDIM) return;
    int o  = idx / KDIM;
    int kk = idx % KDIM;
    int k  = kk / CIN;
    int c  = kk % CIN;
    wpk[idx] = (_Float16)w[(size_t)o * KDIM + c * 9 + k];
}

// ---------------------------------------------------------------------------
// Offset conv: om[n][oc][p] = conv3x3(x, w_off)[oc] + b_off[oc], oc in 0..26
// ---------------------------------------------------------------------------
__global__ __launch_bounds__(256) void offconv_kernel(const float* __restrict__ x,
                                                      const float* __restrict__ wof,
                                                      const float* __restrict__ bof,
                                                      float* __restrict__ om) {
    __shared__ __align__(16) float xt[18 * 18];
    __shared__ __align__(16) float wt[27 * 9];
    const int n   = blockIdx.z;
    const int ty0 = blockIdx.y * 16, tx0 = blockIdx.x * 16;
    const int t   = threadIdx.x;
    const int ly  = t >> 4, lx = t & 15;
    const int gy  = ty0 + ly, gx = tx0 + lx;
    const float* xb = x + (size_t)n * CIN * HW;

    float acc[27];
#pragma unroll
    for (int i = 0; i < 27; ++i) acc[i] = 0.f;

    for (int c = 0; c < CIN; ++c) {
        for (int i = t; i < 324; i += 256) {
            int py = ty0 - 1 + i / 18;
            int px = tx0 - 1 + i % 18;
            float v = 0.f;
            if (py >= 0 && py < Hdim && px >= 0 && px < Wdim)
                v = xb[(size_t)c * HW + py * Wdim + px];
            xt[i] = v;
        }
        if (t < 243) {
            int oc = t / 9, tap = t % 9;
            wt[t] = wof[(size_t)oc * KDIM + c * 9 + tap];
        }
        __syncthreads();
        float xv[9];
#pragma unroll
        for (int tap = 0; tap < 9; ++tap)
            xv[tap] = xt[(ly + tap / 3) * 18 + (lx + tap % 3)];
#pragma unroll
        for (int oc = 0; oc < 27; ++oc) {
            float s = acc[oc];
#pragma unroll
            for (int tap = 0; tap < 9; ++tap) s = fmaf(wt[oc * 9 + tap], xv[tap], s);
            acc[oc] = s;
        }
        __syncthreads();
    }
    int p = gy * Wdim + gx;
    for (int oc = 0; oc < 27; ++oc)
        om[((size_t)n * 27 + oc) * HW + p] = acc[oc] + bof[oc];
}

// ---------------------------------------------------------------------------
// Sampling (one image): smp[p][k*256+c] = (f16)( mask * bilinear(x[c], py, px) )
// One thread per pixel; corner validity folded into the bilinear weights.
// Channels processed in pairs -> 32-bit packed stores.
// ---------------------------------------------------------------------------
__global__ __launch_bounds__(256) void sample_kernel(const float* __restrict__ x,   // [256][96][96]
                                                     const float* __restrict__ om,  // [27][9216]
                                                     _Float16* __restrict__ smp) {  // [9216][2304]
    const int p  = blockIdx.x * 256 + threadIdx.x;     // exact: 36*256 = 9216
    const int pyi = p / Wdim, pxi = p % Wdim;
    _Float16* orow = smp + (size_t)p * KDIM;

    for (int k = 0; k < 9; ++k) {
        float dy = om[(size_t)k * HW + p];
        float dx = om[(size_t)(9 + k) * HW + p];
        float mv = om[(size_t)(18 + k) * HW + p];
        float m  = 1.0f / (1.0f + __expf(-mv));
        float fy = dy + (float)(pyi + k / 3 - 1);
        float fx = dx + (float)(pxi + k % 3 - 1);
        float y0f = floorf(fy), x0f = floorf(fx);
        float lyf = fy - y0f, lxf = fx - x0f;
        int y0 = (int)y0f, x0 = (int)x0f;
        int y1 = y0 + 1,   x1 = x0 + 1;

        float vy0 = (y0 >= 0 && y0 < Hdim) ? 1.f : 0.f;
        float vy1 = (y1 >= 0 && y1 < Hdim) ? 1.f : 0.f;
        float vx0 = (x0 >= 0 && x0 < Wdim) ? 1.f : 0.f;
        float vx1 = (x1 >= 0 && x1 < Wdim) ? 1.f : 0.f;
        int cy0 = min(max(y0, 0), Hdim - 1), cy1 = min(max(y1, 0), Hdim - 1);
        int cx0 = min(max(x0, 0), Wdim - 1), cx1 = min(max(x1, 0), Wdim - 1);

        float w00 = (1.f - lyf) * (1.f - lxf) * vy0 * vx0 * m;
        float w01 = (1.f - lyf) * lxf         * vy0 * vx1 * m;
        float w10 = lyf * (1.f - lxf)         * vy1 * vx0 * m;
        float w11 = lyf * lxf                 * vy1 * vx1 * m;
        int i00 = cy0 * Wdim + cx0, i01 = cy0 * Wdim + cx1;
        int i10 = cy1 * Wdim + cx0, i11 = cy1 * Wdim + cx1;

        _Float16* okc = orow + k * CIN;
        const float* xc0 = x;
        for (int c = 0; c < CIN; c += 2) {
            const float* xa = xc0;
            const float* xb = xc0 + HW;
            float va = w00 * xa[i00] + w01 * xa[i01] + w10 * xa[i10] + w11 * xa[i11];
            float vb = w00 * xb[i00] + w01 * xb[i01] + w10 * xb[i10] + w11 * xb[i11];
            v2h pk; pk[0] = (_Float16)va; pk[1] = (_Float16)vb;
            *(v2h*)(okc + c) = pk;
            xc0 += 2 * HW;
        }
    }
}

// ---------------------------------------------------------------------------
// WMMA GEMM (one image): out[o][p] = bias[o] + sum_kk wpk[o][kk] * smp[p][kk]
// Block: 256 threads = 8 waves; tile 128(o) x 64(p); K-step 32.
// Double-buffered LDS filled by async global->LDS b128; wave -> 2x2 WMMA tiles.
// ---------------------------------------------------------------------------
#define TM 128
#define TN 64
#define TK 32

__global__ __launch_bounds__(256) void gemm_kernel(const _Float16* __restrict__ wpk, // [256][2304]
                                                   const _Float16* __restrict__ smp, // [9216][2304]
                                                   const float* __restrict__ bias,
                                                   float* __restrict__ out) {        // [256][9216]
    __shared__ __align__(16) _Float16 lA[2][TM * TK];  // 2 x 8 KB
    __shared__ __align__(16) _Float16 lB[2][TN * TK];  // 2 x 4 KB

    const int t    = threadIdx.x;
    const int o0   = blockIdx.y * TM;
    const int p0   = blockIdx.x * TN;
    const int wv   = t >> 5;
    const int lane = t & 31;
    const int hi   = lane >> 4;      // half-wave select
    const int l16  = lane & 15;
    const int wm   = (wv & 3) * 32;  // wave's M offset within block tile
    const int wn   = (wv >> 2) * 32; // wave's N offset

    v8f acc[2][2] = {};

    // Stage one K-step tile (A: 512 x b128, 2/thread; B: 256 x b128, 1/thread)
    auto stage = [&](int buf, int k0) {
#pragma unroll
        for (int i = 0; i < 2; ++i) {
            int q   = t + i * 256;
            int row = q >> 2;
            int col = (q & 3) * 8;   // halves
            async_ld128((unsigned)(size_t)&lA[buf][q * 8],
                        wpk + (size_t)(o0 + row) * KDIM + k0 + col);
        }
        {
            int row = t >> 2;
            int col = (t & 3) * 8;
            async_ld128((unsigned)(size_t)&lB[buf][t * 8],
                        smp + (size_t)(p0 + row) * KDIM + k0 + col);
        }
    };

    stage(0, 0);
    wait_async0();
    __syncthreads();

    const int NIT = KDIM / TK;   // 72
    for (int it = 0; it < NIT; ++it) {
        const int cur = it & 1;
        if (it + 1 < NIT) stage(cur ^ 1, (it + 1) * TK);

        // A fragments: lane holds row M=l16; halves [hi*8..+8) and [16+hi*8..+8)
        v16h af[2], bf[2];
#pragma unroll
        for (int mt = 0; mt < 2; ++mt) {
            const _Float16* base = &lA[cur][(wm + mt * 16 + l16) * TK];
            v8h lo = *(const v8h*)(base + hi * 8);
            v8h hh = *(const v8h*)(base + 16 + hi * 8);
#pragma unroll
            for (int i = 0; i < 8; ++i) { af[mt][i] = lo[i]; af[mt][i + 8] = hh[i]; }
        }
        // B fragments: lane holds column N=l16; halves [hi*8..+8) and [16+hi*8..+8)
#pragma unroll
        for (int nt = 0; nt < 2; ++nt) {
            const _Float16* base = &lB[cur][(wn + nt * 16 + l16) * TK];
            v8h lo = *(const v8h*)(base + hi * 8);
            v8h hh = *(const v8h*)(base + 16 + hi * 8);
#pragma unroll
            for (int i = 0; i < 8; ++i) { bf[nt][i] = lo[i]; bf[nt][i + 8] = hh[i]; }
        }
#pragma unroll
        for (int mt = 0; mt < 2; ++mt)
#pragma unroll
            for (int nt = 0; nt < 2; ++nt)
                acc[mt][nt] = __builtin_amdgcn_wmma_f32_16x16x32_f16(
                    false, af[mt], false, bf[nt], (short)0, acc[mt][nt], false, false);

        // Next buffer's async fills must be complete (per-wave), and all waves
        // must be done reading `cur` before it is refilled -> wait + barrier.
        wait_async0();
        __syncthreads();
    }

    // Write-out: C/D layout — lane = N(l16) + 16*(m>=8 half); VGPR r -> M = r + hi*8
#pragma unroll
    for (int mt = 0; mt < 2; ++mt)
#pragma unroll
        for (int nt = 0; nt < 2; ++nt) {
            int ob = o0 + wm + mt * 16 + hi * 8;
            int pb = p0 + wn + nt * 16 + l16;
#pragma unroll
            for (int r = 0; r < 8; ++r) {
                int o = ob + r;
                out[(size_t)o * HW + pb] = acc[mt][nt][r] + bias[o];
            }
        }
}

// ---------------------------------------------------------------------------
// BatchNorm stats: one block per channel, reduce over N*H*W = 36864 elements.
// ---------------------------------------------------------------------------
__global__ __launch_bounds__(256) void bn_stats_kernel(const float* __restrict__ y,
                                                       float* __restrict__ mr) { // [256][2]
    const int o = blockIdx.x;
    const int t = threadIdx.x;
    float s = 0.f, s2 = 0.f;
    for (int i = t; i < NIMG * HW; i += 256) {
        int n = i / HW, p = i % HW;
        float v = y[((size_t)n * COUT + o) * HW + p];
        s += v; s2 += v * v;
    }
    __shared__ float rs[256], rq[256];
    rs[t] = s; rq[t] = s2;
    __syncthreads();
    for (int off = 128; off > 0; off >>= 1) {
        if (t < off) { rs[t] += rs[t + off]; rq[t] += rq[t + off]; }
        __syncthreads();
    }
    if (t == 0) {
        float mean = rs[0] / (float)(NIMG * HW);
        float var  = rq[0] / (float)(NIMG * HW) - mean * mean;
        mr[o * 2 + 0] = mean;
        mr[o * 2 + 1] = rsqrtf(var + 1e-5f);
    }
}

__global__ __launch_bounds__(256) void bn_apply_kernel(float* __restrict__ y,
                                                       const float* __restrict__ mr,
                                                       const float* __restrict__ gamma,
                                                       const float* __restrict__ beta) {
    size_t i = (size_t)blockIdx.x * 256 + threadIdx.x;   // exact grid
    int o = (int)((i / HW) & (COUT - 1));
    float v = y[i];
    v = (v - mr[o * 2]) * mr[o * 2 + 1] * gamma[o] + beta[o];
    y[i] = v > 0.f ? v : 0.f;
}

// ---------------------------------------------------------------------------
extern "C" void kernel_launch(void* const* d_in, const int* in_sizes, int n_in,
                              void* d_out, int out_size, void* d_ws, size_t ws_size,
                              hipStream_t stream) {
    const float* x     = (const float*)d_in[0];
    const float* w_off = (const float*)d_in[1];
    const float* b_off = (const float*)d_in[2];
    const float* w     = (const float*)d_in[3];
    const float* b     = (const float*)d_in[4];
    const float* gamma = (const float*)d_in[5];
    const float* beta  = (const float*)d_in[6];
    float* out = (float*)d_out;

    char* ws = (char*)d_ws;
    float*    om  = (float*)ws;                                 // 3.98 MB
    _Float16* wpk = (_Float16*)(ws + (size_t)4  * 1024 * 1024); // 1.18 MB
    _Float16* smp = (_Float16*)(ws + (size_t)6  * 1024 * 1024); // 42.5 MB
    float*    mr  = (float*)(ws + (size_t)50 * 1024 * 1024);    // 2 KB

    pack_w_kernel<<<(COUT * KDIM + 255) / 256, 256, 0, stream>>>(w, wpk);
    offconv_kernel<<<dim3(6, 6, NIMG), 256, 0, stream>>>(x, w_off, b_off, om);

    for (int n = 0; n < NIMG; ++n) {
        const float* xn = x + (size_t)n * CIN * HW;
        sample_kernel<<<HW / 256, 256, 0, stream>>>(xn, om + (size_t)n * 27 * HW, smp);
        gemm_kernel<<<dim3(HW / TN, COUT / TM), 256, 0, stream>>>(
            wpk, smp, b, out + (size_t)n * COUT * HW);
    }

    bn_stats_kernel<<<COUT, 256, 0, stream>>>(out, mr);
    bn_apply_kernel<<<(NIMG * COUT * HW) / 256, 256, 0, stream>>>(out, mr, gamma, beta);
}
